// RadixLinearAttention_52355651338519
// MI455X (gfx1250) — matrix operand order repaired and checked by reading, not compiled, not run on hardware
//
#include <hip/hip_runtime.h>
#include <hip/hip_bf16.h>

#define Bsz 2
#define Tlen 2048
#define HKh 8
#define HVh 16
#define DKd 128
#define DVd 128
#define KDIM (HKh*DKd)      // 1024
#define VDIM (HVh*DVd)      // 2048
#define CONVD (2*KDIM+VDIM) // 4096
#define KSZ 4
#define EPSv 1e-6f
#define CHUNK 16
#define NCHUNK (Tlen/CHUNK)

typedef __bf16 bf16;
typedef __attribute__((ext_vector_type(16))) __bf16 v16bf;
typedef __attribute__((ext_vector_type(8)))  __bf16 v8bf;
typedef __attribute__((ext_vector_type(4)))  __bf16 v4bf;
typedef __attribute__((ext_vector_type(8)))  float  v8f;

__device__ __forceinline__ v16bf mk16(v8bf lo, v8bf hi) {
  v16bf r;
#pragma unroll
  for (int i = 0; i < 8; ++i) { r[i] = lo[i]; r[8 + i] = hi[i]; }
  return r;
}
__device__ __forceinline__ v8bf zero8() {
  v8bf z;
#pragma unroll
  for (int i = 0; i < 8; ++i) z[i] = (bf16)0.f;
  return z;
}

// ---- CDNA5 async global->LDS copy (ASYNCcnt), 16B per lane ----
__device__ __forceinline__ void async_b128(unsigned int lds_byte_addr, const bf16* gsrc) {
  asm volatile("global_load_async_to_lds_b128 %0, %1, off"
               :: "v"(lds_byte_addr), "v"(gsrc)
               : "memory");
}
__device__ __forceinline__ void wait_async0() {
  asm volatile("s_wait_asynccnt 0" ::: "memory");
}

// ---- WMMA fragment loads: every lane reads contiguous 16B runs (ds_load_b128) ----
__device__ __forceinline__ v16bf a_frag32(const bf16* base, int ld, int k0) {
  int lane = threadIdx.x & 31;
  const v8bf* row = (const v8bf*)(base + (lane & 15) * ld + k0);
  int s = (lane & 16) ? 1 : 0;
  return mk16(row[s], row[s + 2]);
}
__device__ __forceinline__ v16bf a_frag16(const bf16* base, int ld) {
  int lane = threadIdx.x & 31;
  const v8bf* row = (const v8bf*)(base + (lane & 15) * ld);
  return mk16(row[(lane & 16) ? 1 : 0], zero8());
}
__device__ __forceinline__ v16bf b_frag32_cm(const bf16* baseT, int ldT, int k0, int n0) {
  int lane = threadIdx.x & 31;
  const v8bf* col = (const v8bf*)(baseT + (n0 + (lane & 15)) * ldT + k0 + ((lane & 16) ? 8 : 0));
  return mk16(col[0], col[2]);
}
__device__ __forceinline__ v16bf b_frag16_cm(const bf16* baseT, int n0, const bf16* zpad) {
  int lane = threadIdx.x & 31;
  const bf16* src = baseT + (n0 + (lane & 15)) * 16 + ((lane & 16) ? 8 : 0);
  const v8bf* p = (const v8bf*)src;
  return mk16(p[0], zero8());
}

#define WMMA_BF16(Afrag, Bfrag, Cacc) \
  __builtin_amdgcn_wmma_f32_16x16x32_bf16(false, (Afrag), false, (Bfrag), (short)0, (Cacc), false, false)

// ---------------- Kernel 1: conv + SiLU + L2norm + gates -> ws ----------------
__global__ void gdn_preprocess(const float* __restrict__ mixed,
                               const float* __restrict__ a_in,
                               const float* __restrict__ b_in,
                               const float* __restrict__ convw,
                               const float* __restrict__ convb,
                               const float* __restrict__ A_log,
                               const float* __restrict__ dt_bias,
                               void* __restrict__ ws) {
  bf16* qws = (bf16*)ws;
  bf16* kws = qws + (size_t)Bsz * Tlen * HKh * DKd;
  bf16* vws = kws + (size_t)Bsz * Tlen * HKh * DKd;
  float* gws = (float*)(vws + (size_t)Bsz * Tlen * HVh * DVd);
  float* bws = gws + (size_t)Bsz * Tlen * HVh;

  int bt = blockIdx.x;
  int b = bt / Tlen, t = bt % Tlen;
  int tid = threadIdx.x;

  __shared__ float ybuf[CONVD];
  __shared__ float scal[16];

  // conv + SiLU, float4-vectorized (convw is [C][4] -> one float4 per channel)
#pragma unroll
  for (int i = 0; i < 4; ++i) {
    int c = (tid + i * 256) * 4;
    float4 yb = ((const float4*)convb)[c >> 2];
    float yy[4] = {yb.x, yb.y, yb.z, yb.w};
    float4 wv[4];
#pragma unroll
    for (int cc = 0; cc < 4; ++cc) wv[cc] = ((const float4*)convw)[c + cc];
    const float* wf = (const float*)wv;
#pragma unroll
    for (int j = 0; j < KSZ; ++j) {
      int tt = t - (KSZ - 1) + j;
      if (tt >= 0) {
        float4 x = *(const float4*)(mixed + ((size_t)b * Tlen + tt) * CONVD + c);
        yy[0] += x.x * wf[0 * 4 + j];
        yy[1] += x.y * wf[1 * 4 + j];
        yy[2] += x.z * wf[2 * 4 + j];
        yy[3] += x.w * wf[3 * 4 + j];
      }
    }
    float4 o;
    o.x = yy[0] / (1.f + __expf(-yy[0]));
    o.y = yy[1] / (1.f + __expf(-yy[1]));
    o.z = yy[2] / (1.f + __expf(-yy[2]));
    o.w = yy[3] / (1.f + __expf(-yy[3]));
    *(float4*)(ybuf + c) = o;
  }
  __syncthreads();
  if (tid < 16) {  // groups 0..7 = q heads, 8..15 = k heads
    float ss = 0.f;
    const float4* g4 = (const float4*)(ybuf + tid * 128);
    for (int i = 0; i < 32; ++i) {
      float4 x = g4[i];
      ss += x.x * x.x + x.y * x.y + x.z * x.z + x.w * x.w;
    }
    scal[tid] = rsqrtf(ss + EPSv);
  }
  __syncthreads();
  const float qscale = 0.08838834764831845f;  // 128^-0.5
  size_t rowqk = ((size_t)b * Tlen + t) * HKh * DKd;
  size_t rowv  = ((size_t)b * Tlen + t) * HVh * DVd;
  {
    int c = tid * 4;  // 0..1023
    int hq = c >> 7;
    float s0 = scal[hq] * qscale, s1 = scal[8 + hq];
    v4bf qp, kp;
#pragma unroll
    for (int cc = 0; cc < 4; ++cc) {
      qp[cc] = (bf16)(ybuf[c + cc] * s0);
      kp[cc] = (bf16)(ybuf[KDIM + c + cc] * s1);
    }
    *(v4bf*)(qws + rowqk + c) = qp;
    *(v4bf*)(kws + rowqk + c) = kp;
  }
#pragma unroll
  for (int i = 0; i < 2; ++i) {
    int c = tid * 4 + i * 1024;
    v4bf vp;
#pragma unroll
    for (int cc = 0; cc < 4; ++cc) vp[cc] = (bf16)ybuf[2 * KDIM + c + cc];
    *(v4bf*)(vws + rowv + c) = vp;
  }
  if (tid < HVh) {
    int h = tid;
    size_t idx = ((size_t)b * Tlen + t) * HVh + h;
    float x = a_in[idx] + dt_bias[h];
    float sp = (x > 20.f) ? x : log1pf(__expf(x));
    gws[idx] = -__expf(A_log[h]) * sp;
    bws[idx] = 1.f / (1.f + __expf(-b_in[idx]));
  }
}

// ---------------- Kernel 2: chunked gated delta-rule scan (WMMA + async DMA) ----------------
__global__ void __launch_bounds__(256, 1)
gdn_scan(void* __restrict__ ws, float* __restrict__ out) {
  const bf16* qws = (const bf16*)ws;
  const bf16* kws = qws + (size_t)Bsz * Tlen * HKh * DKd;
  const bf16* vws = kws + (size_t)Bsz * Tlen * HKh * DKd;
  const float* gws = (const float*)(vws + (size_t)Bsz * Tlen * HVh * DVd);
  const float* bws = gws + (size_t)Bsz * Tlen * HVh;

  int b = blockIdx.x / HVh, h = blockIdx.x % HVh, hq = h >> 1;  // rep = HV/HK = 2
  int tid = threadIdx.x, lane = tid & 31, w = tid >> 5;

  __shared__ bf16 SbT[128 * 128];         // state, COLUMN-major: SbT[n*128 + k]
  __shared__ bf16 Kb[2][CHUNK * DKd];     // raw k chunk (double-buffered, async DMA)
  __shared__ bf16 Qb[2][CHUNK * DKd];     // raw q chunk (double-buffered, async DMA)
  __shared__ bf16 Vb[2][CHUNK * DVd];     // raw v chunk (double-buffered, async DMA)
  __shared__ bf16 KTw[DKd * CHUNK];       // (k * D_last/D_s)^T : [d][t]
  __shared__ bf16 XbT[DVd * CHUNK];       // delta^T bf16: [d][t]
  __shared__ bf16 MfB[CHUNK * CHUNK];     // intra-chunk attention (bf16, A operand)
  __shared__ bf16 Zpad[16];               // zero pad for K=16..31 halves
  __shared__ float Xf[CHUNK * DVd];       // V -> U -> delta (f32), row-major [t][d]
  __shared__ float Af[CHUNK * CHUNK];     // strictly-lower A (f32, for fwd subst)
  __shared__ float Esc[CHUNK], Einv[CHUNK], Eow[CHUNK], Bet[CHUNK];
  __shared__ float Dlast;

  for (int i = tid; i < 128 * 128 / 8; i += 256) ((v8bf*)SbT)[i] = zero8();
  if (tid < 16) Zpad[tid] = (bf16)0.f;
  v8f accS[8];  // wave w owns state rows (DK) [16w,16w+16), all 8 DV tiles (fp32)
#pragma unroll
  for (int n = 0; n < 8; ++n) accS[n] = (v8f){};

  const int m0 = w * 16, n0 = w * 16;
  const int rbase = (lane & 16) ? 8 : 0;
  const int cN = lane & 15;
  const int ltt = (tid * 8) >> 7, ld0 = (tid * 8) & 127;  // 256 thr x 8 elem = 16x128

  // async-prefetch one chunk's K/Q/V (each thread DMAs its 16B slice of each)
  auto issue = [&](int ch, int buf) {
    size_t rqk = (((size_t)b * Tlen + ch * CHUNK + ltt) * HKh + hq) * (size_t)DKd + ld0;
    size_t rv  = (((size_t)b * Tlen + ch * CHUNK + ltt) * HVh + h) * (size_t)DVd + ld0;
    async_b128((unsigned int)(size_t)(Kb[buf] + ltt * DKd + ld0), kws + rqk);
    async_b128((unsigned int)(size_t)(Qb[buf] + ltt * DKd + ld0), qws + rqk);
    async_b128((unsigned int)(size_t)(Vb[buf] + ltt * DVd + ld0), vws + rv);
  };

  issue(0, 0);
  wait_async0();
  __syncthreads();

  for (int chunk = 0; chunk < NCHUNK; ++chunk) {
    int t0 = chunk * CHUNK;
    int cur = chunk & 1;
    const bf16* KbC = Kb[cur];
    const bf16* QbC = Qb[cur];

    if (chunk + 1 < NCHUNK) issue(chunk + 1, cur ^ 1);  // overlap DMA with compute

    {  // V (bf16, LDS) -> Xf (f32)
      v8bf vv = *(const v8bf*)(Vb[cur] + ltt * DVd + ld0);
      float4 f0, f1;
      f0.x = (float)vv[0]; f0.y = (float)vv[1]; f0.z = (float)vv[2]; f0.w = (float)vv[3];
      f1.x = (float)vv[4]; f1.y = (float)vv[5]; f1.z = (float)vv[6]; f1.w = (float)vv[7];
      float4* xf4 = (float4*)(Xf + ltt * DVd + ld0);
      xf4[0] = f0; xf4[1] = f1;
    }
    if (tid == 0) {  // chunk decay scalars
      float G = 0.f, Gt[CHUNK];
      for (int tt = 0; tt < CHUNK; ++tt) {
        size_t idx = ((size_t)b * Tlen + t0 + tt) * HVh + h;
        G += gws[idx]; Gt[tt] = G;
        Esc[tt] = __expf(G); Einv[tt] = __expf(-G);
        Bet[tt] = bws[idx];
      }
      for (int tt = 0; tt < CHUNK; ++tt) Eow[tt] = __expf(G - Gt[tt]);
      Dlast = Esc[CHUNK - 1];
    }
    __syncthreads();

    // KS = K @ S_old (tile (0,w)); U = beta_t * (V - D_t * KS)
    {
      v8f acc = {};
#pragma unroll
      for (int kt = 0; kt < 4; ++kt)
        acc = WMMA_BF16(a_frag32(KbC, DKd, kt * 32),
                        b_frag32_cm(SbT, 128, kt * 32, n0), acc);
#pragma unroll
      for (int r = 0; r < 8; ++r) {
        int row = r + rbase;
        Xf[row * DVd + n0 + cN] =
            Bet[row] * (Xf[row * DVd + n0 + cN] - Esc[row] * acc[r]);
      }
    }
    if (w == 0) {  // P = K K^T -> A[t][s] = beta_t * P * D_t/D_s  (s<t)
      v8f p = {};
#pragma unroll
      for (int kt = 0; kt < 4; ++kt)
        p = WMMA_BF16(a_frag32(KbC, DKd, kt * 32), b_frag32_cm(KbC, DKd, kt * 32, 0), p);
#pragma unroll
      for (int r = 0; r < 8; ++r) {
        int row = r + rbase;
        Af[row * 16 + cN] = (cN < row) ? Bet[row] * p[r] * Esc[row] * Einv[cN] : 0.f;
      }
    }
    if (w == 1) {  // P2 = Q K^T -> M[t][s] = P2 * D_t/D_s  (s<=t), bf16
      v8f p = {};
#pragma unroll
      for (int kt = 0; kt < 4; ++kt)
        p = WMMA_BF16(a_frag32(QbC, DKd, kt * 32), b_frag32_cm(KbC, DKd, kt * 32, 0), p);
#pragma unroll
      for (int r = 0; r < 8; ++r) {
        int row = r + rbase;
        MfB[row * 16 + cN] = (bf16)((cN <= row) ? p[r] * Esc[row] * Einv[cN] : 0.f);
      }
    }
    __syncthreads();
    // forward substitution: delta = (I + A)^{-1} U  (16x16 triangular over 128 cols)
    for (int t = 1; t < CHUNK; ++t) {
      if (tid < DVd) {
        float s = 0.f;
        for (int ss2 = 0; ss2 < t; ++ss2) s += Af[t * 16 + ss2] * Xf[ss2 * DVd + tid];
        Xf[t * DVd + tid] -= s;
      }
      __syncthreads();
    }
    // build transposed bf16 operands: delta^T and (K * D_last/D_s)^T
    for (int i = tid; i < CHUNK * DKd; i += 256) {
      int tt = i >> 7, d = i & 127;
      XbT[d * 16 + tt] = (bf16)Xf[tt * DVd + d];
      KTw[d * 16 + tt] = (bf16)((float)KbC[tt * DKd + d] * Eow[tt]);
    }
    __syncthreads();

    // O = D_t * (Q @ S_old) + M @ delta   (tile (0,w)) -> global
    {
      v8f o = {};
#pragma unroll
      for (int kt = 0; kt < 4; ++kt)
        o = WMMA_BF16(a_frag32(QbC, DKd, kt * 32),
                      b_frag32_cm(SbT, 128, kt * 32, n0), o);
#pragma unroll
      for (int r = 0; r < 8; ++r) o[r] *= Esc[r + rbase];  // row scale by D_t
      o = WMMA_BF16(a_frag16(MfB, 16), b_frag16_cm(XbT, n0, Zpad), o);
#pragma unroll
      for (int r = 0; r < 8; ++r)
        out[((size_t)b * Tlen + t0 + r + rbase) * VDIM + h * DVd + n0 + cN] = o[r];
    }
    // S_new = D_last * S_old + (K * D_last/D_s)^T @ delta   (wave w: DK-tile w)
#pragma unroll
    for (int nt = 0; nt < 8; ++nt) {
      v8f c = accS[nt];
#pragma unroll
      for (int j = 0; j < 8; ++j) c[j] *= Dlast;
      c = WMMA_BF16(a_frag16(KTw + m0 * 16, 16), b_frag16_cm(XbT, nt * 16, Zpad), c);
      accS[nt] = c;
    }
    __syncthreads();  // everyone done reading SbT before overwrite
#pragma unroll
    for (int nt = 0; nt < 8; ++nt) {
      v8bf pk;
#pragma unroll
      for (int r = 0; r < 8; ++r) pk[r] = (bf16)accS[nt][r];
      *(v8bf*)(SbT + (nt * 16 + cN) * 128 + m0 + rbase) = pk;  // 16B packed store
    }
    wait_async0();    // prefetched chunk resident in LDS
    __syncthreads();  // make it (and SbT) visible block-wide
  }
}

extern "C" void kernel_launch(void* const* d_in, const int* in_sizes, int n_in,
                              void* d_out, int out_size, void* d_ws, size_t ws_size,
                              hipStream_t stream) {
  (void)in_sizes; (void)n_in; (void)out_size; (void)ws_size;
  const float* mixed   = (const float*)d_in[0];
  const float* a_in    = (const float*)d_in[1];
  const float* b_in    = (const float*)d_in[2];
  const float* convw   = (const float*)d_in[3];
  const float* convb   = (const float*)d_in[4];
  const float* A_log   = (const float*)d_in[5];
  const float* dt_bias = (const float*)d_in[6];
  float* out = (float*)d_out;

  gdn_preprocess<<<Bsz * Tlen, 256, 0, stream>>>(mixed, a_in, b_in, convw, convb,
                                                 A_log, dt_bias, d_ws);
  gdn_scan<<<Bsz * HVh, 256, 0, stream>>>(d_ws, out);
}
